// PC_Trs_20212116095320
// MI455X (gfx1250) — matrix operand order repaired and checked by reading, not compiled, not run on hardware
//
#include <hip/hip_runtime.h>
#include <math.h>

#define BB 4
#define NN 2048
#define PP (BB * NN)
#define NSAMP 64
#define KPTS 15
#define CKP 64
#define FF 256
#define HH 8
#define DHH 32
#define HIDN 512
#define LL 2
#define RAD2 0.04f            // RADIUS^2 = 0.2^2
#define KPE_INV 12.5f         // 1 / KP_EXTENT = 1/0.08
#define WIN_INV 4.0f          // 1 / WINDOW = 1/0.25
#define LN_EPS 1e-5f
#define ATTN_SCALE 0.17677669529663689f  // 1/sqrt(32)

typedef __attribute__((ext_vector_type(16))) _Float16 v16h;
typedef __attribute__((ext_vector_type(8)))  _Float16 v8h;
typedef __attribute__((ext_vector_type(8)))  float    v8f;

#define CAT16(lo, hi) __builtin_shufflevector((lo), (hi), 0, 1, 2, 3, 4, 5, 6, 7, \
                                              8, 9, 10, 11, 12, 13, 14, 15)

// ---------------------------------------------------------------------------
// 1) Ball query: first NSAMP in-radius neighbors per point, ascending index,
//    fill = PP  (matches reference stable-argsort semantics)
// ---------------------------------------------------------------------------
__global__ __launch_bounds__(256) void ball_query_kernel(
    const float* __restrict__ pos, int* __restrict__ idx)
{
  int p = blockIdx.x * 256 + threadIdx.x;
  if (p >= PP) return;
  int b = p / NN;
  float px = pos[p * 3 + 0], py = pos[p * 3 + 1], pz = pos[p * 3 + 2];
  const float* bp = pos + (size_t)b * NN * 3;
  int* op = idx + (size_t)p * NSAMP;
  int cnt = 0;
  for (int j = 0; j < NN && cnt < NSAMP; ++j) {
    float dx = bp[j * 3 + 0] - px;
    float dy = bp[j * 3 + 1] - py;
    float dz = bp[j * 3 + 2] - pz;
    float d2 = dx * dx + dy * dy + dz * dz;
    if (d2 <= RAD2) op[cnt++] = b * NN + j;
  }
  for (; cnt < NSAMP; ++cnt) op[cnt] = PP;
}

// ---------------------------------------------------------------------------
// 2) KPConv (raw, pre-BN)
// ---------------------------------------------------------------------------
__global__ __launch_bounds__(256) void kpconv_kernel(
    const float* __restrict__ pos, const float* __restrict__ feat,
    const int* __restrict__ idx, const float* __restrict__ kp,
    const float* __restrict__ kpw, float* __restrict__ outraw)
{
  __shared__ float kps[KPTS * 3];
  __shared__ float wsh[KPTS * 3 * CKP];
  int t = threadIdx.x;
  for (int i = t; i < KPTS * 3; i += 256) kps[i] = kp[i];
  for (int i = t; i < KPTS * 3 * CKP; i += 256) wsh[i] = kpw[i];
  __syncthreads();

  int p = blockIdx.x * 256 + t;
  if (p >= PP) return;
  float px = pos[p * 3 + 0], py = pos[p * 3 + 1], pz = pos[p * 3 + 2];

  float wf[KPTS][3];
#pragma unroll
  for (int k = 0; k < KPTS; ++k) { wf[k][0] = 0.f; wf[k][1] = 0.f; wf[k][2] = 0.f; }

  const int* ip = idx + (size_t)p * NSAMP;
  for (int ns = 0; ns < NSAMP; ++ns) {
    int id = ip[ns];
    if (id >= PP) break;  // trailing fill entries contribute zero
    float nx = pos[id * 3 + 0] - px;
    float ny = pos[id * 3 + 1] - py;
    float nz = pos[id * 3 + 2] - pz;
    float f0 = feat[id * 3 + 0], f1 = feat[id * 3 + 1], f2 = feat[id * 3 + 2];
#pragma unroll
    for (int k = 0; k < KPTS; ++k) {
      float dx = nx - kps[k * 3 + 0];
      float dy = ny - kps[k * 3 + 1];
      float dz = nz - kps[k * 3 + 2];
      float dist = sqrtf(dx * dx + dy * dy + dz * dz + 1e-12f);
      float infl = fmaxf(1.f - dist * KPE_INV, 0.f);
      wf[k][0] += infl * f0;
      wf[k][1] += infl * f1;
      wf[k][2] += infl * f2;
    }
  }
  float* op = outraw + (size_t)p * CKP;
  for (int c = 0; c < CKP; ++c) {
    float acc = 0.f;
#pragma unroll
    for (int k = 0; k < KPTS; ++k) {
      acc += wf[k][0] * wsh[(k * 3 + 0) * CKP + c];
      acc += wf[k][1] * wsh[(k * 3 + 1) * CKP + c];
      acc += wf[k][2] * wsh[(k * 3 + 2) * CKP + c];
    }
    op[c] = acc;
  }
}

// ---------------------------------------------------------------------------
// 3) BatchNorm over P points (per channel) + LeakyReLU(0.2)
// ---------------------------------------------------------------------------
__global__ __launch_bounds__(256) void bn_stats_kernel(
    const float* __restrict__ x, float* __restrict__ stats)
{
  __shared__ float s_sum[256], s_sq[256];
  int c = blockIdx.x, t = threadIdx.x;
  float sum = 0.f, sq = 0.f;
  for (int p = t; p < PP; p += 256) {
    float v = x[(size_t)p * CKP + c];
    sum += v; sq += v * v;
  }
  s_sum[t] = sum; s_sq[t] = sq;
  __syncthreads();
  for (int s = 128; s > 0; s >>= 1) {
    if (t < s) { s_sum[t] += s_sum[t + s]; s_sq[t] += s_sq[t + s]; }
    __syncthreads();
  }
  if (t == 0) { stats[c] = s_sum[0]; stats[CKP + c] = s_sq[0]; }
}

__global__ __launch_bounds__(256) void bn_apply_kernel(
    const float* __restrict__ x, const float* __restrict__ stats,
    const float* __restrict__ gamma, const float* __restrict__ beta,
    float* __restrict__ y)
{
  int i = blockIdx.x * 256 + threadIdx.x;   // PP*CKP elements
  int c = i & (CKP - 1);
  float mu  = stats[c] * (1.f / PP);
  float var = stats[CKP + c] * (1.f / PP) - mu * mu;
  float v = gamma[c] * (x[i] - mu) * rsqrtf(var + LN_EPS) + beta[c];
  y[i] = (v >= 0.f) ? v : 0.2f * v;
}

// ---------------------------------------------------------------------------
// 4) Voxel ids
// ---------------------------------------------------------------------------
__global__ __launch_bounds__(256) void bmin_kernel(
    const float* __restrict__ pos, float* __restrict__ bmin)
{
  __shared__ float red[256][3];
  int b = blockIdx.x, t = threadIdx.x;
  float mn[3] = {3.0e38f, 3.0e38f, 3.0e38f};
  const float* bp = pos + (size_t)b * NN * 3;
  for (int j = t; j < NN; j += 256)
    for (int d = 0; d < 3; ++d) mn[d] = fminf(mn[d], bp[j * 3 + d]);
  red[t][0] = mn[0]; red[t][1] = mn[1]; red[t][2] = mn[2];
  __syncthreads();
  for (int s = 128; s > 0; s >>= 1) {
    if (t < s)
      for (int d = 0; d < 3; ++d) red[t][d] = fminf(red[t][d], red[t + s][d]);
    __syncthreads();
  }
  if (t == 0)
    for (int d = 0; d < 3; ++d) bmin[b * 3 + d] = red[0][d];
}

__global__ __launch_bounds__(256) void cid_kernel(
    const float* __restrict__ pos, const float* __restrict__ bmin,
    int* __restrict__ cid)
{
  int p = blockIdx.x * 256 + threadIdx.x;
  if (p >= PP) return;
  int b = p / NN;
  int cx = (int)floorf((pos[p * 3 + 0] - bmin[b * 3 + 0]) * WIN_INV);
  int cy = (int)floorf((pos[p * 3 + 1] - bmin[b * 3 + 1]) * WIN_INV);
  int cz = (int)floorf((pos[p * 3 + 2] - bmin[b * 3 + 2]) * WIN_INV);
  cid[p] = cx * 4096 + cy * 64 + cz;
}

// ---------------------------------------------------------------------------
// 5) Embedding: X[p][f] = [pos(3) | kpf(64)] @ Wm[67][256]   (tiny K -> scalar)
// ---------------------------------------------------------------------------
__global__ __launch_bounds__(256) void embed_kernel(
    const float* __restrict__ pos, const float* __restrict__ kpf,
    const float* __restrict__ Wm, float* __restrict__ X)
{
  int id = blockIdx.x * 256 + threadIdx.x;  // PP*FF
  int p = id >> 8, f = id & (FF - 1);
  float acc = 0.f;
#pragma unroll
  for (int c = 0; c < 3; ++c) acc += pos[p * 3 + c] * Wm[c * FF + f];
  for (int c = 0; c < CKP; ++c) acc += kpf[(size_t)p * CKP + c] * Wm[(c + 3) * FF + f];
  X[id] = acc;
}

// ---------------------------------------------------------------------------
// 6a) f32 -> f16 flat convert (activations)
// ---------------------------------------------------------------------------
__global__ __launch_bounds__(256) void cvt_f16_kernel(
    const float* __restrict__ in, _Float16* __restrict__ out)
{
  int i = blockIdx.x * 256 + threadIdx.x;
  out[i] = (_Float16)in[i];
}

// 6b) weight [K][N] f32 -> transposed [N][K] f16
__global__ __launch_bounds__(256) void cvt_wt_kernel(
    const float* __restrict__ W, _Float16* __restrict__ Wt, int K, int N)
{
  int id = blockIdx.x * 256 + threadIdx.x;   // over N*K outputs
  int n = id / K, k = id - n * K;
  Wt[id] = (_Float16)W[(size_t)k * N + n];
}

// ---------------------------------------------------------------------------
// 6c) LDS-free WMMA GEMM: C[M,N] = Ah[M,K] @ Bt[N,K]^T + bias (+resid)(+relu)
//     Block = 256 thr, 128x64 tile; each wave owns a 32x32 quadrant
//     -> 4 x v_wmma_f32_16x16x32_f16 per K-step, operands via global b128.
// ---------------------------------------------------------------------------
__global__ __launch_bounds__(256) void gemm_f16_kernel(
    const _Float16* __restrict__ Ah, const _Float16* __restrict__ Bt,
    const float* __restrict__ bias, const float* __restrict__ resid,
    float* __restrict__ C, int M, int K, int N, int relu)
{
  int t = threadIdx.x, wave = t >> 5, lane = t & 31;
  int ln16 = lane & 15, half = lane >> 4;
  int bm = blockIdx.y * 128, bn = blockIdx.x * 64;
  int qr = wave >> 1, qc = wave & 1;        // 4x2 wave grid of 32x32 quadrants

  int ar0 = bm + qr * 32 + ln16;            // A fragment rows (two subtiles)
  int ar1 = ar0 + 16;
  int bc0 = bn + qc * 32 + ln16;            // B fragment cols (two subtiles)
  int bc1 = bc0 + 16;

  v8f acc00 = {}, acc01 = {}, acc10 = {}, acc11 = {};
  for (int k0 = 0; k0 < K; k0 += 32) {
    const v8h* pa0 = (const v8h*)(Ah + (size_t)ar0 * K + k0);
    const v8h* pa1 = (const v8h*)(Ah + (size_t)ar1 * K + k0);
    const v8h* pb0 = (const v8h*)(Bt + (size_t)bc0 * K + k0);
    const v8h* pb1 = (const v8h*)(Bt + (size_t)bc1 * K + k0);
    if (k0 + 32 < K) {
      __builtin_prefetch((const _Float16*)pa0 + 32, 0, 1);  // global_prefetch_b8
      __builtin_prefetch((const _Float16*)pb0 + 32, 0, 1);
    }
    // A layout per lane: K = {koff..koff+7, 16+koff..16+koff+7}, koff = half*8
    v16h a0 = CAT16(pa0[half], pa0[2 + half]);
    v16h a1 = CAT16(pa1[half], pa1[2 + half]);
    // B layout per lane: K = {koff..koff+7, 16+koff..16+koff+7}, koff = half*8
    v16h b0 = CAT16(pb0[half], pb0[2 + half]);
    v16h b1 = CAT16(pb1[half], pb1[2 + half]);

    acc00 = __builtin_amdgcn_wmma_f32_16x16x32_f16(false, a0, false, b0,
                                                   (short)0, acc00, false, false);
    acc01 = __builtin_amdgcn_wmma_f32_16x16x32_f16(false, a0, false, b1,
                                                   (short)0, acc01, false, false);
    acc10 = __builtin_amdgcn_wmma_f32_16x16x32_f16(false, a1, false, b0,
                                                   (short)0, acc10, false, false);
    acc11 = __builtin_amdgcn_wmma_f32_16x16x32_f16(false, a1, false, b1,
                                                   (short)0, acc11, false, false);
  }

  int orow = bm + qr * 32 + half * 8;       // C layout: row = 8*half + r
  int ocol = bn + qc * 32 + ln16;
  float bi0 = bias[ocol], bi1 = bias[ocol + 16];
#pragma unroll
  for (int r = 0; r < 8; ++r) {
    float v00 = acc00[r] + bi0, v01 = acc01[r] + bi1;
    float v10 = acc10[r] + bi0, v11 = acc11[r] + bi1;
    if (resid) {
      v00 += resid[(size_t)(orow + r) * N + ocol];
      v01 += resid[(size_t)(orow + r) * N + ocol + 16];
      v10 += resid[(size_t)(orow + 16 + r) * N + ocol];
      v11 += resid[(size_t)(orow + 16 + r) * N + ocol + 16];
    }
    if (relu) {
      v00 = fmaxf(v00, 0.f); v01 = fmaxf(v01, 0.f);
      v10 = fmaxf(v10, 0.f); v11 = fmaxf(v11, 0.f);
    }
    C[(size_t)(orow + r) * N + ocol]           = v00;
    C[(size_t)(orow + r) * N + ocol + 16]      = v01;
    C[(size_t)(orow + 16 + r) * N + ocol]      = v10;
    C[(size_t)(orow + 16 + r) * N + ocol + 16] = v11;
  }
}

// ---------------------------------------------------------------------------
// 7) Flash attention with voxel mask (vectorized LDS fragments)
// ---------------------------------------------------------------------------
__global__ __launch_bounds__(256) void attn_kernel(
    const float* __restrict__ Q, const float* __restrict__ K,
    const float* __restrict__ V, const int* __restrict__ cid,
    float* __restrict__ O)
{
  __shared__ _Float16 q_s[64][40];   // row stride 80B (16B-aligned)
  __shared__ _Float16 k_s[64][40];
  __shared__ _Float16 v_s[64][40];
  __shared__ float    s_s[64][68];
  __shared__ _Float16 p_s[64][72];   // row stride 144B (16B-aligned)
  __shared__ float m_s[64], l_s[64], f_s[64];
  __shared__ int cq_s[64], ck_s[64];

  int qt = blockIdx.x, h = blockIdx.y, b = blockIdx.z;
  int t = threadIdx.x, wave = t >> 5, lane = t & 31;
  int ln16 = lane & 15, half = lane >> 4;
  int koff = half * 8, kb = half * 16;
  int otr = wave >> 1, otc = wave & 1;   // 4x2 grid of 16x16 output subtiles

  int qrow0 = b * NN + qt * 64;
  {
    int e = t * 8, r = e >> 5, c0 = e & 31;
    const float* sq = Q + (size_t)(qrow0 + r) * FF + h * DHH + c0;
    v8h qv;
#pragma unroll
    for (int i = 0; i < 8; ++i) qv[i] = (_Float16)sq[i];
    *(v8h*)&q_s[r][c0] = qv;   // single ds_store_b128
  }
  if (t < 64) {
    cq_s[t] = cid[qrow0 + t];
    m_s[t] = -3.0e38f;
    l_s[t] = 0.f;
  }
  v8f acc = {};
  __syncthreads();

  for (int kt = 0; kt < NN / 64; ++kt) {
    int krow0 = b * NN + kt * 64;
    {
      int e = t * 8, r = e >> 5, c0 = e & 31;
      const float* sk = K + (size_t)(krow0 + r) * FF + h * DHH + c0;
      const float* sv = V + (size_t)(krow0 + r) * FF + h * DHH + c0;
      v8h kv, vv;
#pragma unroll
      for (int i = 0; i < 8; ++i) { kv[i] = (_Float16)sk[i]; vv[i] = (_Float16)sv[i]; }
      *(v8h*)&k_s[r][c0] = kv;
      *(v8h*)&v_s[r][c0] = vv;
      if (t < 64) ck_s[t] = cid[krow0 + t];
    }
    __syncthreads();

    // --- S = Q @ K^T * scale : 16 subtiles, 2 per wave (K=DH=32 -> 1 WMMA)
#pragma unroll
    for (int s = 0; s < 2; ++s) {
      int st = wave * 2 + s, tr = st >> 2, tc = st & 3;
      int ar = tr * 16 + ln16;
      v16h a = CAT16(*(const v8h*)&q_s[ar][koff],
                     *(const v8h*)&q_s[ar][16 + koff]);
      int bc = tc * 16 + ln16;                   // B[kk][n] = k_s[n][kk]
      v16h bf = CAT16(*(const v8h*)&k_s[bc][koff],
                      *(const v8h*)&k_s[bc][16 + koff]);
      v8f c0 = {};
      c0 = __builtin_amdgcn_wmma_f32_16x16x32_f16(false, a, false, bf,
                                                  (short)0, c0, false, false);
      int sr = tr * 16 + half * 8, sc = tc * 16 + ln16;
#pragma unroll
      for (int r = 0; r < 8; ++r) s_s[sr + r][sc] = c0[r] * ATTN_SCALE;
    }
    __syncthreads();

    // --- masked online softmax (one thread per query row)
    if (t < 64) {
      int row = t, cq = cq_s[row];
      float mold = m_s[row], mnew = mold;
      for (int j = 0; j < 64; ++j) {
        float x = (ck_s[j] == cq) ? s_s[row][j] : -1e9f;
        s_s[row][j] = x;
        mnew = fmaxf(mnew, x);
      }
      float fac = __expf(mold - mnew);
      float ls = 0.f;
      for (int j = 0; j < 64; ++j) {
        float pv = __expf(s_s[row][j] - mnew);
        p_s[row][j] = (_Float16)pv;
        ls += pv;
      }
      m_s[row] = mnew;
      l_s[row] = l_s[row] * fac + ls;
      f_s[row] = fac;
    }
    __syncthreads();

    // --- rescale accumulator, then O += P @ V (K=64 -> two WMMAs)
    int arow = otr * 16 + half * 8;
#pragma unroll
    for (int r = 0; r < 8; ++r) acc[r] *= f_s[arow + r];

    int pr = otr * 16 + ln16;
    v16h a0 = CAT16(*(const v8h*)&p_s[pr][koff],
                    *(const v8h*)&p_s[pr][16 + koff]);
    v16h a1 = CAT16(*(const v8h*)&p_s[pr][32 + koff],
                    *(const v8h*)&p_s[pr][48 + koff]);
    int vc = otc * 16 + ln16;
    v16h b0, b1;                      // inherent transpose -> strided (scalar)
#pragma unroll
    for (int i = 0; i < 16; ++i) {
      b0[i] = v_s[((i & 8) << 1) + koff + (i & 7)][vc];
      b1[i] = v_s[32 + ((i & 8) << 1) + koff + (i & 7)][vc];
    }
    acc = __builtin_amdgcn_wmma_f32_16x16x32_f16(false, a0, false, b0,
                                                 (short)0, acc, false, false);
    acc = __builtin_amdgcn_wmma_f32_16x16x32_f16(false, a1, false, b1,
                                                 (short)0, acc, false, false);
    __syncthreads();
  }

  int orow = otr * 16 + half * 8;
  int ocol = h * DHH + otc * 16 + ln16;
#pragma unroll
  for (int r = 0; r < 8; ++r)
    O[(size_t)(qrow0 + orow + r) * FF + ocol] = acc[r] / l_s[orow + r];
}

// ---------------------------------------------------------------------------
// 8) LayerNorm over F=256: one wave32 per row
// ---------------------------------------------------------------------------
__global__ __launch_bounds__(256) void ln_kernel(
    const float* __restrict__ x, const float* __restrict__ g,
    const float* __restrict__ be, float* __restrict__ out)
{
  int row = blockIdx.x * 8 + (threadIdx.x >> 5);
  int lane = threadIdx.x & 31;
  const float* xr = x + (size_t)row * FF;
  float v[8], sum = 0.f, sq = 0.f;
#pragma unroll
  for (int i = 0; i < 8; ++i) {
    v[i] = xr[lane + 32 * i];
    sum += v[i];
    sq += v[i] * v[i];
  }
#pragma unroll
  for (int off = 16; off > 0; off >>= 1) {
    sum += __shfl_xor(sum, off, 32);
    sq  += __shfl_xor(sq, off, 32);
  }
  float mu = sum * (1.f / FF);
  float var = sq * (1.f / FF) - mu * mu;
  float rs = rsqrtf(var + LN_EPS);
#pragma unroll
  for (int i = 0; i < 8; ++i) {
    int c = lane + 32 * i;
    out[(size_t)row * FF + c] = g[c] * (v[i] - mu) * rs + be[c];
  }
}

// ---------------------------------------------------------------------------
// Host orchestration
// ---------------------------------------------------------------------------
extern "C" void kernel_launch(void* const* d_in, const int* in_sizes, int n_in,
                              void* d_out, int out_size, void* d_ws, size_t ws_size,
                              hipStream_t stream) {
  (void)in_sizes; (void)n_in; (void)out_size; (void)ws_size;
  const float* pos   = (const float*)d_in[0];
  const float* feat  = (const float*)d_in[1];
  const float* kp    = (const float*)d_in[2];
  const float* kpw   = (const float*)d_in[3];
  const float* bng   = (const float*)d_in[4];
  const float* bnb   = (const float*)d_in[5];
  const float* Wm    = (const float*)d_in[6];
  const float* Wq    = (const float*)d_in[7];
  const float* bq    = (const float*)d_in[8];
  const float* Wk    = (const float*)d_in[9];
  const float* bk    = (const float*)d_in[10];
  const float* Wv    = (const float*)d_in[11];
  const float* bv    = (const float*)d_in[12];
  const float* Wo    = (const float*)d_in[13];
  const float* bo    = (const float*)d_in[14];
  const float* g1    = (const float*)d_in[15];
  const float* be1   = (const float*)d_in[16];
  const float* W1    = (const float*)d_in[17];
  const float* b1    = (const float*)d_in[18];
  const float* W2    = (const float*)d_in[19];
  const float* b2    = (const float*)d_in[20];
  const float* g2    = (const float*)d_in[21];
  const float* be2   = (const float*)d_in[22];
  float* outp = (float*)d_out;

  // workspace bump allocator
  char* w = (char*)d_ws;
  auto alloc = [&](size_t bytes) -> void* {
    void* p = (void*)w;
    w += (bytes + 255) & ~(size_t)255;
    return p;
  };
  int*      idx   = (int*)     alloc((size_t)PP * NSAMP * 4);
  float*    kpraw = (float*)   alloc((size_t)PP * CKP * 4);
  float*    stats = (float*)   alloc(2 * CKP * 4);
  float*    kpf   = (float*)   alloc((size_t)PP * CKP * 4);
  float*    bmn   = (float*)   alloc(BB * 3 * 4);
  int*      cidb  = (int*)     alloc((size_t)PP * 4);
  float*    X     = (float*)   alloc((size_t)PP * FF * 4);
  float*    Qb    = (float*)   alloc((size_t)PP * FF * 4);
  float*    Kb    = (float*)   alloc((size_t)PP * FF * 4);
  float*    Vb    = (float*)   alloc((size_t)PP * FF * 4);
  float*    Ob    = (float*)   alloc((size_t)PP * FF * 4);
  float*    T1    = (float*)   alloc((size_t)PP * FF * 4);
  float*    X1    = (float*)   alloc((size_t)PP * FF * 4);
  float*    Hb    = (float*)   alloc((size_t)PP * HIDN * 4);
  float*    T2    = (float*)   alloc((size_t)PP * FF * 4);
  _Float16* Ah    = (_Float16*)alloc((size_t)PP * FF * 2);    // f16 activations
  _Float16* Hh    = (_Float16*)alloc((size_t)PP * HIDN * 2);
  _Float16* Wt    = (_Float16*)alloc((size_t)HIDN * FF * 2);  // f16 weight^T

  // --- geometry + KPConv front-end
  ball_query_kernel<<<PP / 256, 256, 0, stream>>>(pos, idx);
  kpconv_kernel<<<PP / 256, 256, 0, stream>>>(pos, feat, idx, kp, kpw, kpraw);
  bn_stats_kernel<<<CKP, 256, 0, stream>>>(kpraw, stats);
  bn_apply_kernel<<<(PP * CKP) / 256, 256, 0, stream>>>(kpraw, stats, bng, bnb, kpf);
  bmin_kernel<<<BB, 256, 0, stream>>>(pos, bmn);
  cid_kernel<<<PP / 256, 256, 0, stream>>>(pos, bmn, cidb);
  embed_kernel<<<(PP * FF) / 256, 256, 0, stream>>>(pos, kpf, Wm, X);

  // --- transformer encoder layers
  const int nXF = (PP * FF) / 256, nXH = (PP * HIDN) / 256;
  const int nWF = (FF * FF) / 256, nWH = (FF * HIDN) / 256;
  dim3 gF(FF / 64, PP / 128);      // (4,64)
  dim3 gH(HIDN / 64, PP / 128);    // (8,64)
  dim3 gA(NN / 64, HH, BB);        // (32,8,4)

  for (int l = 0; l < LL; ++l) {
    const float* wq = Wq + (size_t)l * FF * FF;
    const float* wk = Wk + (size_t)l * FF * FF;
    const float* wv = Wv + (size_t)l * FF * FF;
    const float* wo = Wo + (size_t)l * FF * FF;
    const float* w1 = W1 + (size_t)l * FF * HIDN;
    const float* w2 = W2 + (size_t)l * HIDN * FF;

    cvt_f16_kernel<<<nXF, 256, 0, stream>>>(X, Ah);
    cvt_wt_kernel<<<nWF, 256, 0, stream>>>(wq, Wt, FF, FF);
    gemm_f16_kernel<<<gF, 256, 0, stream>>>(Ah, Wt, bq + l * FF, nullptr, Qb, PP, FF, FF, 0);
    cvt_wt_kernel<<<nWF, 256, 0, stream>>>(wk, Wt, FF, FF);
    gemm_f16_kernel<<<gF, 256, 0, stream>>>(Ah, Wt, bk + l * FF, nullptr, Kb, PP, FF, FF, 0);
    cvt_wt_kernel<<<nWF, 256, 0, stream>>>(wv, Wt, FF, FF);
    gemm_f16_kernel<<<gF, 256, 0, stream>>>(Ah, Wt, bv + l * FF, nullptr, Vb, PP, FF, FF, 0);

    attn_kernel<<<gA, 256, 0, stream>>>(Qb, Kb, Vb, cidb, Ob);

    cvt_f16_kernel<<<nXF, 256, 0, stream>>>(Ob, Ah);
    cvt_wt_kernel<<<nWF, 256, 0, stream>>>(wo, Wt, FF, FF);
    gemm_f16_kernel<<<gF, 256, 0, stream>>>(Ah, Wt, bo + l * FF, X, T1, PP, FF, FF, 0);
    ln_kernel<<<PP / 8, 256, 0, stream>>>(T1, g1 + l * FF, be1 + l * FF, X1);

    cvt_f16_kernel<<<nXF, 256, 0, stream>>>(X1, Ah);
    cvt_wt_kernel<<<nWH, 256, 0, stream>>>(w1, Wt, FF, HIDN);
    gemm_f16_kernel<<<gH, 256, 0, stream>>>(Ah, Wt, b1 + l * HIDN, nullptr, Hb, PP, FF, HIDN, 1);

    cvt_f16_kernel<<<nXH, 256, 0, stream>>>(Hb, Hh);
    cvt_wt_kernel<<<nWH, 256, 0, stream>>>(w2, Wt, HIDN, FF);
    gemm_f16_kernel<<<gF, 256, 0, stream>>>(Hh, Wt, b2 + l * FF, X1, T2, PP, HIDN, FF, 0);

    float* dst = (l == LL - 1) ? outp : X;
    ln_kernel<<<PP / 8, 256, 0, stream>>>(T2, g2 + l * FF, be2 + l * FF, dst);
  }
}
